// HierarchicalTransformer_9320079032571
// MI455X (gfx1250) — matrix operand order repaired and checked by reading, not compiled, not run on hardware
//
#include <hip/hip_runtime.h>
#include <hip/hip_bf16.h>
#include <math.h>

typedef __attribute__((ext_vector_type(16))) _Float16 v16h;
typedef __attribute__((ext_vector_type(8)))  _Float16 v8h;
typedef __attribute__((ext_vector_type(4)))  _Float16 v4h;
typedef __attribute__((ext_vector_type(8)))  float    v8f;
typedef __attribute__((ext_vector_type(4)))  float    v4f;

#define D_MODEL 1024
#define FFDIM   4096
#define NHEAD   16
#define DK      64
#define SEQ     4096
#define NBATCH  4
#define NROWS   (NBATCH*SEQ)      /* 16384 token rows */
#define D2      (D_MODEL*D_MODEL)

__device__ __forceinline__ float waveSum(float v) {
#pragma unroll
  for (int o = 16; o > 0; o >>= 1) v += __shfl_xor(v, o, 32);
  return v;
}

// ---------------------------------------------------------------------------
// Embedding gather + clip + sinusoidal positional encoding
// ---------------------------------------------------------------------------
__global__ void k_embed(const int* __restrict__ x, const float* __restrict__ emb,
                        float* __restrict__ out) {
  int idx = blockIdx.x * blockDim.x + threadIdx.x;   // over NROWS*D_MODEL
  int row = idx >> 10;
  int col = idx & (D_MODEL - 1);
  int s   = row & (SEQ - 1);
  int tok = x[row];
  float e = emb[(size_t)tok * D_MODEL + col];
  e = fminf(fmaxf(e, -100.f), 100.f);
  int j = col >> 1;
  float div = expf((float)(2 * j) * (-9.210340371976184f / (float)D_MODEL));
  float ang = (float)s * div;
  float pe  = (col & 1) ? cosf(ang) : sinf(ang);
  out[idx] = e + pe;
}

// ---------------------------------------------------------------------------
// LayerNorm over D_MODEL, one row per workgroup (256 threads, 8 waves).
// out = LN(in)*w + b  [+ res]  [clip to +-100]
// ---------------------------------------------------------------------------
__global__ void k_layernorm(const float* __restrict__ in, const float* __restrict__ w,
                            const float* __restrict__ b, const float* __restrict__ res,
                            float* __restrict__ out, int clip) {
  int row = blockIdx.x;
  int t   = threadIdx.x;                       // 0..255
  size_t base = (size_t)row * D_MODEL;
  const float* xr = in + base;
  float v0 = xr[t], v1 = xr[t + 256], v2 = xr[t + 512], v3 = xr[t + 768];

  __shared__ float sw[8];
  __shared__ float stat;

  float s = waveSum(v0 + v1 + v2 + v3);
  if ((t & 31) == 0) sw[t >> 5] = s;
  __syncthreads();
  if (t == 0) { float tot = 0.f; for (int i = 0; i < 8; ++i) tot += sw[i]; stat = tot / (float)D_MODEL; }
  __syncthreads();
  float m = stat;
  float d0 = v0 - m, d1 = v1 - m, d2 = v2 - m, d3 = v3 - m;
  float q = waveSum(d0 * d0 + d1 * d1 + d2 * d2 + d3 * d3);
  __syncthreads();
  if ((t & 31) == 0) sw[t >> 5] = q;
  __syncthreads();
  if (t == 0) { float tot = 0.f; for (int i = 0; i < 8; ++i) tot += sw[i]; stat = tot / (float)D_MODEL; }
  __syncthreads();
  float rstd = rsqrtf(stat + 1e-5f);

  float dv[4] = { d0, d1, d2, d3 };
#pragma unroll
  for (int i = 0; i < 4; ++i) {
    int c = t + i * 256;
    float o = dv[i] * rstd * w[c] + b[c];
    if (res) o += res[base + c];
    if (clip) o = fminf(fmaxf(o, -100.f), 100.f);
    out[base + c] = o;
  }
}

// ---------------------------------------------------------------------------
// WMMA GEMM:  C[M,N] = op( A[M,K] @ W[K,N] + bias [+ res] )
// 128x128 tile per 128-thread workgroup (4 waves, 64x64 per wave), BK=32.
// Double-buffered LDS: tile k+1 is fetched (global_load_b128, f32) into
// registers while 16 v_wmma_f32_16x16x32_f16 consume tile k from LDS.
// op: 0 = none, 1 = exact GELU, 2 = + residual.
// Requires M%128==0, N%128==0, K%32==0 (true for every GEMM in this model).
// ---------------------------------------------------------------------------
__global__ __launch_bounds__(128)
void k_gemm_f16(const float* __restrict__ A, const float* __restrict__ W,
                const float* __restrict__ bias, const float* __restrict__ res,
                float* __restrict__ C, int N, int K, int op) {
  // row stride 40 halves (80 B): every fragment chunk stays 16B-aligned.
  __shared__ __align__(16) _Float16 sA[2][128 * 40];
  __shared__ __align__(16) _Float16 sB[2][128 * 40];   // stored [n][k]

  const int tid  = threadIdx.x;
  const int wave = tid >> 5;
  const int lane = tid & 31;
  const int hi   = lane >> 4;          // half-wave select
  const int lm   = lane & 15;
  const int wr   = (wave & 1) * 64;    // wave row offset inside tile
  const int wc   = (wave >> 1) * 64;   // wave col offset inside tile
  const int tM   = blockIdx.y * 128;
  const int tN   = blockIdx.x * 128;

  v8f acc[4][4];
#pragma unroll
  for (int p = 0; p < 4; ++p)
#pragma unroll
    for (int j = 0; j < 4; ++j)
#pragma unroll
      for (int r = 0; r < 8; ++r) acc[p][j][r] = 0.f;

  v4f ra[8], rb[8];                    // register staging for next tile

  auto loadTile = [&](int k0) {        // global -> registers (vector loads)
#pragma unroll
    for (int i = 0; i < 8; ++i) {      // A tile: 128 rows x 32 K
      int e = tid + i * 128;           // 0..1023 float4 chunks
      int r = e >> 3, c4 = e & 7;
      ra[i] = *(const v4f*)&A[(size_t)(tM + r) * K + k0 + c4 * 4];
    }
#pragma unroll
    for (int i = 0; i < 8; ++i) {      // B tile: 32 K x 128 N
      int e = tid + i * 128;
      int kk = e >> 5, n4 = e & 31;
      rb[i] = *(const v4f*)&W[(size_t)(k0 + kk) * N + tN + n4 * 4];
    }
  };
  auto storeTile = [&](int buf) {      // registers -> LDS (f32 -> f16)
#pragma unroll
    for (int i = 0; i < 8; ++i) {
      int e = tid + i * 128;
      int r = e >> 3, c4 = e & 7;
      v4h h;
#pragma unroll
      for (int j = 0; j < 4; ++j) h[j] = (_Float16)ra[i][j];
      *(v4h*)&sA[buf][r * 40 + c4 * 4] = h;     // ds_store_b64
    }
#pragma unroll
    for (int i = 0; i < 8; ++i) {                // transpose B on store
      int e = tid + i * 128;
      int kk = e >> 5, n4 = e & 31;
#pragma unroll
      for (int j = 0; j < 4; ++j)
        sB[buf][(n4 * 4 + j) * 40 + kk] = (_Float16)rb[i][j];
    }
  };

  const int KT = K >> 5;
  loadTile(0);
  storeTile(0);
  __syncthreads();

  for (int kt = 0; kt < KT; ++kt) {
    const int cur = kt & 1;
    if (kt + 1 < KT) loadTile((kt + 1) << 5);    // prefetch next tile

    v16h af[4], bf[4];
#pragma unroll
    for (int p = 0; p < 4; ++p) {
      // A frag: half 0 -> K{0..7,16..23}, half 1 -> K{8..15,24..31}
      int row = wr + p * 16 + lm;
      v8h lo = *(const v8h*)&sA[cur][row * 40 + hi * 8];
      v8h hh = *(const v8h*)&sA[cur][row * 40 + 16 + hi * 8];
#pragma unroll
      for (int u = 0; u < 8; ++u) { af[p][u] = lo[u]; af[p][u + 8] = hh[u]; }
    }
#pragma unroll
    for (int j = 0; j < 4; ++j) {
      // B frag: half 0 -> K{0..7,16..23}, half 1 -> K{8..15,24..31} of column
      int n = wc + j * 16 + lm;
      v8h lo = *(const v8h*)&sB[cur][n * 40 + hi * 8];
      v8h hh = *(const v8h*)&sB[cur][n * 40 + 16 + hi * 8];
#pragma unroll
      for (int u = 0; u < 8; ++u) { bf[j][u] = lo[u]; bf[j][u + 8] = hh[u]; }
    }
#pragma unroll
    for (int p = 0; p < 4; ++p)
#pragma unroll
      for (int j = 0; j < 4; ++j)
        acc[p][j] = __builtin_amdgcn_wmma_f32_16x16x32_f16(
            false, af[p], false, bf[j], (short)0, acc[p][j], false, false);

    if (kt + 1 < KT) storeTile(cur ^ 1);         // fill the other buffer
    __syncthreads();
  }

  // C/D layout: VGPR r -> row (hi*8 + r), lane lm -> col
#pragma unroll
  for (int p = 0; p < 4; ++p) {
#pragma unroll
    for (int j = 0; j < 4; ++j) {
      int col = tN + wc + j * 16 + lm;
      float bb = bias[col];
#pragma unroll
      for (int r = 0; r < 8; ++r) {
        int row = tM + wr + p * 16 + hi * 8 + r;
        float v = acc[p][j][r] + bb;
        if (op == 1)      v = 0.5f * v * (1.f + erff(v * 0.70710678118654752f));
        else if (op == 2) v += res[(size_t)row * N + col];
        C[(size_t)row * N + col] = v;
      }
    }
  }
}

// ---------------------------------------------------------------------------
// Block attention, one workgroup per (block, head). blockDim.x == S (<=128).
// Exact max-subtract softmax via online (flash) recurrence; q pre-scaled.
// ---------------------------------------------------------------------------
__global__ void k_attn(const float* __restrict__ Q, const float* __restrict__ K,
                       const float* __restrict__ V, float* __restrict__ O, int S) {
  __shared__ float sK[128 * DK];
  __shared__ float sV[128 * DK];
  int blk  = blockIdx.x >> 4;
  int head = blockIdx.x & 15;
  int t    = threadIdx.x;                 // query index within block
  size_t rb = (size_t)blk * S;
  size_t cb = (size_t)head * DK;
#pragma unroll
  for (int d = 0; d < DK; ++d) {
    sK[t * DK + d] = K[(rb + t) * D_MODEL + cb + d];
    sV[t * DK + d] = V[(rb + t) * D_MODEL + cb + d];
  }
  __syncthreads();

  float qr[DK];
#pragma unroll
  for (int d = 0; d < DK; ++d) qr[d] = Q[(rb + t) * D_MODEL + cb + d] * 0.125f;

  float m = -3.0e38f, l = 0.f, av[DK];
#pragma unroll
  for (int d = 0; d < DK; ++d) av[d] = 0.f;

  for (int s = 0; s < S; ++s) {
    float dot = 0.f;
#pragma unroll
    for (int d = 0; d < DK; ++d) dot += qr[d] * sK[s * DK + d];
    float nm = fmaxf(m, dot);
    float f  = expf(m - nm);
    float p  = expf(dot - nm);
    l = l * f + p;
#pragma unroll
    for (int d = 0; d < DK; ++d) av[d] = av[d] * f + p * sV[s * DK + d];
    m = nm;
  }
  float inv = 1.f / (l + 1e-9f);
#pragma unroll
  for (int d = 0; d < DK; ++d) O[(rb + t) * D_MODEL + cb + d] = av[d] * inv;
}

// ---------------------------------------------------------------------------
// Attention pooling: single shared query vector against each 128-row block.
// Materializes softmax -> applies exact clip(a, 1e-9, 1) like the reference.
// ---------------------------------------------------------------------------
__global__ void k_pool_attn(const float* __restrict__ qv, const float* __restrict__ K,
                            const float* __restrict__ V, float* __restrict__ O) {
  __shared__ float sc[NHEAD][128];
  int blk = blockIdx.x;
  int t   = threadIdx.x;                  // key index 0..127
  size_t rb = (size_t)blk * 128;
  for (int h = 0; h < NHEAD; ++h) {
    const float* kr = K + (rb + t) * D_MODEL + h * DK;
    const float* qh = qv + h * DK;
    float dot = 0.f;
#pragma unroll
    for (int d = 0; d < DK; ++d) dot += qh[d] * kr[d];
    sc[h][t] = dot * 0.125f;
  }
  __syncthreads();
  if (t < NHEAD) {
    float mx = -3.0e38f;
    for (int s = 0; s < 128; ++s) mx = fmaxf(mx, sc[t][s]);
    float sum = 0.f;
    for (int s = 0; s < 128; ++s) sum += expf(sc[t][s] - mx);
    float inv = 1.f / (sum + 1e-9f);
    for (int s = 0; s < 128; ++s) {
      float a = expf(sc[t][s] - mx) * inv;
      sc[t][s] = fminf(fmaxf(a, 1e-9f), 1.f);
    }
  }
  __syncthreads();
#pragma unroll
  for (int i = 0; i < 8; ++i) {
    int c = t + i * 128;
    int h = c >> 6;
    float o = 0.f;
    for (int s = 0; s < 128; ++s) o += sc[h][s] * V[(rb + s) * D_MODEL + c];
    O[(size_t)blk * D_MODEL + c] = o;
  }
}

// out[n] = bias[n] + sum_k a[k] * W[k*N + n]   (single-row projection)
__global__ void k_matvec(const float* __restrict__ a, const float* __restrict__ W,
                         const float* __restrict__ bias, float* __restrict__ out,
                         int Kd, int N) {
  int n = blockIdx.x * blockDim.x + threadIdx.x;
  if (n >= N) return;
  float s = bias[n];
  for (int k = 0; k < Kd; ++k) s += a[k] * W[(size_t)k * N + n];
  out[n] = s;
}

// ---------------------------------------------------------------------------
extern "C" void kernel_launch(void* const* d_in, const int* in_sizes, int n_in,
                              void* d_out, int out_size, void* d_ws, size_t ws_size,
                              hipStream_t stream) {
  (void)in_sizes; (void)n_in; (void)out_size; (void)ws_size;

  const int*   x        = (const int*)  d_in[0];
  const float* emb      = (const float*)d_in[1];
  const float* in_ln_w  = (const float*)d_in[2];
  const float* in_ln_b  = (const float*)d_in[3];
  const float* loc_ln_w = (const float*)d_in[4];
  const float* loc_ln_b = (const float*)d_in[5];
  const float* loc_pw   = (const float*)d_in[6];
  const float* loc_pb   = (const float*)d_in[7];
  const float* loc_w1   = (const float*)d_in[8];
  const float* loc_b1   = (const float*)d_in[9];
  const float* loc_w2   = (const float*)d_in[10];
  const float* loc_b2   = (const float*)d_in[11];
  const float* locn_w   = (const float*)d_in[12];
  const float* locn_b   = (const float*)d_in[13];
  const float* pool_q   = (const float*)d_in[14];
  const float* pool_n_w = (const float*)d_in[15];
  const float* pool_n_b = (const float*)d_in[16];
  const float* pool_ln_w= (const float*)d_in[17];
  const float* pool_ln_b= (const float*)d_in[18];
  const float* pool_pw  = (const float*)d_in[19];
  const float* pool_pb  = (const float*)d_in[20];
  const float* glob_ln_w= (const float*)d_in[21];
  const float* glob_ln_b= (const float*)d_in[22];
  const float* glob_pw  = (const float*)d_in[23];
  const float* glob_pb  = (const float*)d_in[24];
  const float* glob_w1  = (const float*)d_in[25];
  const float* glob_b1  = (const float*)d_in[26];
  const float* glob_w2  = (const float*)d_in[27];
  const float* glob_b2  = (const float*)d_in[28];
  const float* out_ln_w = (const float*)d_in[29];
  const float* out_ln_b = (const float*)d_in[30];
  const float* out_w    = (const float*)d_in[31];
  const float* out_b    = (const float*)d_in[32];

  float* ws = (float*)d_ws;
  const size_t F = (size_t)NROWS * D_MODEL;   // 16M floats
  float* H    = ws;                            // activations
  float* T    = ws + 1 * F;                    // LN scratch / GEMM temp
  float* Qb   = ws + 2 * F;
  float* Kb   = ws + 3 * F;
  float* Vb   = ws + 4 * F;
  float* Ob   = ws + 5 * F;
  float* FF   = Qb;                            // FFN hidden aliases Q..O (dead there)
  float* G    = ws + 6 * F;                    // 128 x D pooled tokens
  float* G2   = G  + 128 * D_MODEL;
  float* qtmp = G2 + 128 * D_MODEL;            // LN(pool_q)
  float* qvec = qtmp + D_MODEL;                // projected pooled query

  auto ln = [&](const float* in, const float* w, const float* b, const float* res,
                float* out, int rows, int clip) {
    k_layernorm<<<dim3(rows), dim3(256), 0, stream>>>(in, w, b, res, out, clip);
  };
  auto gemm = [&](const float* A, const float* W, const float* bias, const float* res,
                  float* C, int M, int N, int K, int op) {
    k_gemm_f16<<<dim3(N / 128, M / 128), dim3(128), 0, stream>>>(A, W, bias, res, C, N, K, op);
  };
  // One full transformer layer (pre-LN proj -> attn -> o-proj -> +res -> FFN)
  auto tlayer = [&](const float* lnw, const float* lnb, const float* pw, const float* pb,
                    const float* w1, const float* b1, const float* w2, const float* b2,
                    float* X, int M, int S) {
    int nblk = M / S;
    ln(X, lnw + 0 * D_MODEL, lnb + 0 * D_MODEL, nullptr, T, M, 0);
    gemm(T, pw + 0 * D2, pb + 0 * D_MODEL, nullptr, Qb, M, D_MODEL, D_MODEL, 0);
    ln(X, lnw + 1 * D_MODEL, lnb + 1 * D_MODEL, nullptr, T, M, 0);
    gemm(T, pw + 1 * D2, pb + 1 * D_MODEL, nullptr, Kb, M, D_MODEL, D_MODEL, 0);
    ln(X, lnw + 2 * D_MODEL, lnb + 2 * D_MODEL, nullptr, T, M, 0);
    gemm(T, pw + 2 * D2, pb + 2 * D_MODEL, nullptr, Vb, M, D_MODEL, D_MODEL, 0);
    k_attn<<<dim3(nblk * NHEAD), dim3(S), 0, stream>>>(Qb, Kb, Vb, Ob, S);
    gemm(Ob, pw + 3 * D2, pb + 3 * D_MODEL, nullptr, T, M, D_MODEL, D_MODEL, 0);
    ln(T, lnw + 3 * D_MODEL, lnb + 3 * D_MODEL, X, X, M, 0);      // X = X + LN(o)
    ln(X, lnw + 4 * D_MODEL, lnb + 4 * D_MODEL, nullptr, T, M, 0);
    gemm(T, w1, b1, nullptr, FF, M, FFDIM, D_MODEL, 1);           // GELU
    gemm(FF, w2, b2, X, T, M, D_MODEL, FFDIM, 2);                 // + residual X
    ln(T, lnw + 5 * D_MODEL, lnb + 5 * D_MODEL, nullptr, X, M, 1);// LN + clip
  };

  // --- embedding + input LN ---
  k_embed<<<dim3((NROWS * D_MODEL) / 256), dim3(256), 0, stream>>>(x, emb, H);
  ln(H, in_ln_w, in_ln_b, nullptr, H, NROWS, 0);

  // --- local layers (blocks of 128 tokens) ---
  for (int i = 0; i < 2; ++i) {
    tlayer(loc_ln_w + i * 6 * D_MODEL, loc_ln_b + i * 6 * D_MODEL,
           loc_pw + i * 4 * D2, loc_pb + (size_t)i * 4 * D_MODEL,
           loc_w1 + (size_t)i * D_MODEL * FFDIM, loc_b1 + (size_t)i * FFDIM,
           loc_w2 + (size_t)i * FFDIM * D_MODEL, loc_b2 + (size_t)i * D_MODEL,
           H, NROWS, 128);
  }

  // --- pooling stage ---
  ln(H, locn_w, locn_b, nullptr, H, NROWS, 0);
  ln(H, pool_n_w, pool_n_b, nullptr, T, NROWS, 0);                // hp
  ln(pool_q, pool_ln_w + 0 * D_MODEL, pool_ln_b + 0 * D_MODEL, nullptr, qtmp, 1, 0);
  k_matvec<<<dim3(4), dim3(256), 0, stream>>>(qtmp, pool_pw + 0 * D2,
                                              pool_pb + 0 * D_MODEL, qvec,
                                              D_MODEL, D_MODEL);
  ln(T, pool_ln_w + 1 * D_MODEL, pool_ln_b + 1 * D_MODEL, nullptr, Qb, NROWS, 0);
  gemm(Qb, pool_pw + 1 * D2, pool_pb + 1 * D_MODEL, nullptr, Kb, NROWS, D_MODEL, D_MODEL, 0);
  ln(T, pool_ln_w + 2 * D_MODEL, pool_ln_b + 2 * D_MODEL, nullptr, Qb, NROWS, 0);
  gemm(Qb, pool_pw + 2 * D2, pool_pb + 2 * D_MODEL, nullptr, Vb, NROWS, D_MODEL, D_MODEL, 0);
  k_pool_attn<<<dim3(128), dim3(128), 0, stream>>>(qvec, Kb, Vb, G2);
  gemm(G2, pool_pw + 3 * D2, pool_pb + 3 * D_MODEL, nullptr, G, 128, D_MODEL, D_MODEL, 0);
  ln(G, pool_ln_w + 3 * D_MODEL, pool_ln_b + 3 * D_MODEL, nullptr, G2, 128, 0);

  // --- global layers (4 batches x 32 pooled tokens) ---
  for (int i = 0; i < 2; ++i) {
    tlayer(glob_ln_w + i * 6 * D_MODEL, glob_ln_b + i * 6 * D_MODEL,
           glob_pw + i * 4 * D2, glob_pb + (size_t)i * 4 * D_MODEL,
           glob_w1 + (size_t)i * D_MODEL * FFDIM, glob_b1 + (size_t)i * FFDIM,
           glob_w2 + (size_t)i * FFDIM * D_MODEL, glob_b2 + (size_t)i * D_MODEL,
           G2, 128, 32);
  }

  // --- output head ---
  ln(G2, out_ln_w, out_ln_b, nullptr, T, 128, 0);
  gemm(T, out_w, out_b, nullptr, (float*)d_out, 128, 32000, D_MODEL, 0);
}
